// Block_69810398429476
// MI455X (gfx1250) — compile-verified, matrix operand order checked by reading, not run on hardware
//
#include <hip/hip_runtime.h>
#include <stdint.h>

// ---------------------------------------------------------------------------
// GPT-J style transformer block for MI455X (gfx1250, wave32, WMMA bf16).
// GEMMs: v_wmma_f32_16x16x32_bf16 (fp32 accum), tiles staged to LDS by the
// Tensor Data Mover (tensor_load_to_lds, TENSORcnt) with double buffering.
// ---------------------------------------------------------------------------

#define NTOK   2048   // N
#define DMODEL 2048   // D
#define NH     16     // heads
#define DH     128    // head dim
#define DR     64     // rotary dims
#define QKVC   6144   // 3*H*DH
#define FFD    8192   // 4*D

typedef __attribute__((ext_vector_type(16))) __bf16 v16bf;
typedef __attribute__((ext_vector_type(8)))  float  v8f;
typedef unsigned short ushort_t;

union Frag {
  v16bf v;
  uint4 u[2];
};

__device__ __forceinline__ ushort_t f2bf(float f) {
  unsigned int u = __float_as_uint(f);
  u += 0x7FFFu + ((u >> 16) & 1u);   // round-to-nearest-even
  return (ushort_t)(u >> 16);
}

__device__ __forceinline__ v8f wmma_bf16(v16bf a, v16bf b, v8f c) {
  return __builtin_amdgcn_wmma_f32_16x16x32_bf16(false, a, false, b,
                                                 (short)0, c, false, false);
}

// ---------------------------------------------------------------------------
// Tensor Data Mover staging (CDNA5). One wave issues a D# descriptor; the
// TDM DMAs a 128x32 bf16 tile global->LDS, inserting 4 DWORDs of padding
// every 16 DWORDs so the LDS row stride is 40 ushorts (bank-conflict-free
// 16B fragment reads). Tracked with TENSORcnt.
// ---------------------------------------------------------------------------
#if defined(__gfx1250__) && __has_builtin(__builtin_amdgcn_tensor_load_to_lds)
#define USE_TDM 1
typedef unsigned int u32x4 __attribute__((ext_vector_type(4)));
typedef int          i32x4 __attribute__((ext_vector_type(4)));
typedef int          i32x8 __attribute__((ext_vector_type(8)));

__device__ __forceinline__ void tdm_load_tile(const ushort_t* gptr,
                                              unsigned int lds_off,
                                              int rowLen, int nRows) {
  unsigned long long ga = (unsigned long long)(uintptr_t)gptr;
  u32x4 g0;
  g0[0] = 1u;                                   // count=1, user descriptor
  g0[1] = lds_off;                              // lds_addr (bytes)
  g0[2] = (unsigned int)ga;                     // global_addr[31:0]
  g0[3] = (unsigned int)(ga >> 32) | (2u << 30);// global_addr[56:32] | type=2
  i32x8 g1;
  // data_size=1 (2B), pad_enable=1, pad_interval=3 (16 DW), pad_amount=3 (4 DW)
  g1[0] = (1 << 16) | (1 << 20) | (3 << 22) | (3 << 25);
  g1[1] = (rowLen & 0xFFFF) << 16;                      // tensor_dim0 lo
  g1[2] = (rowLen >> 16) | ((nRows & 0xFFFF) << 16);    // dim0 hi | tensor_dim1 lo
  g1[3] = (nRows >> 16) | (32 << 16);                   // dim1 hi | tile_dim0=32
  g1[4] = 128;                                          // tile_dim1=128, tile_dim2=0
  g1[5] = rowLen;                                       // tensor_dim0_stride lo
  g1[6] = 0;                                            // stride hi | dim1_stride lo
  g1[7] = 0;
  i32x4 z4 = {0, 0, 0, 0};
#if defined(__clang_major__) && (__clang_major__ >= 23)
  i32x8 z8 = {0, 0, 0, 0, 0, 0, 0, 0};
  __builtin_amdgcn_tensor_load_to_lds(g0, g1, z4, z4, z8, 0);
#else
  __builtin_amdgcn_tensor_load_to_lds(g0, g1, z4, z4, 0);
#endif
}
#else
#define USE_TDM 0
#endif

// ---------------------------------------------------------------------------
// LayerNorm: x[N][D] fp32 -> h[N][D] bf16. One block per row, 256 threads.
// ---------------------------------------------------------------------------
__global__ __launch_bounds__(256) void ln_kernel(const float* __restrict__ x,
                                                 const float* __restrict__ sc,
                                                 const float* __restrict__ of,
                                                 ushort_t* __restrict__ hb) {
  int row = blockIdx.x;
  int tid = threadIdx.x;
  const float* xr = x + (size_t)row * DMODEL;
  float v[8];
  float s = 0.f, s2 = 0.f;
#pragma unroll
  for (int i = 0; i < 8; ++i) {
    float t = xr[tid + i * 256];
    v[i] = t; s += t; s2 += t * t;
  }
#pragma unroll
  for (int m = 1; m < 32; m <<= 1) {
    s  += __shfl_xor(s,  m, 32);
    s2 += __shfl_xor(s2, m, 32);
  }
  __shared__ float ps[8], ps2[8];
  int wave = tid >> 5, lane = tid & 31;
  if (lane == 0) { ps[wave] = s; ps2[wave] = s2; }
  __syncthreads();
  s = ps[lane & 7]; s2 = ps2[lane & 7];
#pragma unroll
  for (int m = 1; m < 8; m <<= 1) {
    s  += __shfl_xor(s,  m, 32);
    s2 += __shfl_xor(s2, m, 32);
  }
  float mu  = s * (1.f / DMODEL);
  float var = s2 * (1.f / DMODEL) - mu * mu;
  float inv = rsqrtf(var + 1e-5f);
#pragma unroll
  for (int i = 0; i < 8; ++i) {
    int c = tid + i * 256;
    float hn = (v[i] - mu) * inv * sc[c] + of[c];
    hb[(size_t)row * DMODEL + c] = f2bf(hn);
  }
}

// ---------------------------------------------------------------------------
// Weight transpose + fp32->bf16: W[R][C] -> Wt[C][R] (bf16).
// ---------------------------------------------------------------------------
__global__ __launch_bounds__(256) void wt_kernel(const float* __restrict__ W,
                                                 ushort_t* __restrict__ Wt,
                                                 int R, int C) {
  __shared__ float tile[32][33];
  int bx = blockIdx.x * 32, by = blockIdx.y * 32;
  int tx = threadIdx.x & 31, ty = threadIdx.x >> 5;
  for (int i = ty; i < 32; i += 8)
    tile[i][tx] = W[(size_t)(by + i) * C + bx + tx];
  __syncthreads();
  for (int i = ty; i < 32; i += 8)
    Wt[(size_t)(bx + i) * R + by + tx] = f2bf(tile[tx][i]);
}

// ---------------------------------------------------------------------------
// WMMA GEMM: out[M][Nc] = A[M][K] (bf16 row-major) * BT[Nc][K]^T (bf16).
// Block tile 128x128, KB=32, 256 threads = 8 waves (2x4 wave grid),
// each wave 64x32 = 8 x (16x16) accumulators. TDM double-buffered staging.
// ---------------------------------------------------------------------------
template <int OUTBF, int BIAS, int GELU, int ACC>
__global__ __launch_bounds__(256) void gemm_kernel(
    const ushort_t* __restrict__ A, const ushort_t* __restrict__ BT,
    const float* __restrict__ bias, float* __restrict__ outF,
    ushort_t* __restrict__ outB, int M, int Nc, int K) {
  __shared__ alignas(16) ushort_t As[2][128][40];
  __shared__ alignas(16) ushort_t Bs[2][128][40];
  int tid = threadIdx.x;
  int wave = tid >> 5, lane = tid & 31;
  int wr = wave >> 2, wc = wave & 3;
  int mlane = lane & 15, kb8 = (lane >> 4) * 8;
  int rB = blockIdx.y * 128, cB = blockIdx.x * 128;

  v8f acc[4][2] = {};

#if USE_TDM
  // Prologue: kick off tile 0 into buffer 0 (one wave issues; EXEC ignored).
  unsigned int ldsA[2] = {(unsigned int)(uintptr_t)&As[0][0][0],
                          (unsigned int)(uintptr_t)&As[1][0][0]};
  unsigned int ldsB[2] = {(unsigned int)(uintptr_t)&Bs[0][0][0],
                          (unsigned int)(uintptr_t)&Bs[1][0][0]};
  if (tid == 0) {
    tdm_load_tile(&A[(size_t)rB * K],  ldsA[0], K, M);
    tdm_load_tile(&BT[(size_t)cB * K], ldsB[0], K, Nc);
  }
  for (int k0 = 0; k0 < K; k0 += 32) {
    int buf = (k0 >> 5) & 1;
    if (tid == 0) {
      if (k0 + 32 < K) {
        // Prefetch next tile into the other buffer, then wait so that the
        // two oldest descriptors (current tile) have completed.
        tdm_load_tile(&A[(size_t)rB * K + k0 + 32],  ldsA[buf ^ 1], K, M);
        tdm_load_tile(&BT[(size_t)cB * K + k0 + 32], ldsB[buf ^ 1], K, Nc);
        __builtin_amdgcn_s_wait_tensorcnt(2);
      } else {
        __builtin_amdgcn_s_wait_tensorcnt(0);
      }
    }
    __syncthreads();

    Frag a[4], b[2];
#pragma unroll
    for (int i = 0; i < 4; ++i) {
      const ushort_t* p = &As[buf][wr * 64 + i * 16 + mlane][kb8];
      a[i].u[0] = *(const uint4*)p;
      a[i].u[1] = *(const uint4*)(p + 16);
    }
#pragma unroll
    for (int j = 0; j < 2; ++j) {
      const ushort_t* p = &Bs[buf][wc * 32 + j * 16 + mlane][kb8];
      b[j].u[0] = *(const uint4*)p;
      b[j].u[1] = *(const uint4*)(p + 16);
    }
#pragma unroll
    for (int i = 0; i < 4; ++i)
#pragma unroll
      for (int j = 0; j < 2; ++j)
        acc[i][j] = wmma_bf16(a[i].v, b[j].v, acc[i][j]);
    __syncthreads();  // all waves done with `buf` before TDM overwrites it
  }
#else
  // Fallback: manual staging (no TDM builtin on this toolchain).
  for (int k0 = 0; k0 < K; k0 += 32) {
#pragma unroll
    for (int c = tid; c < 512; c += 256) {
      int r = c >> 2, cc = (c & 3) * 8;
      *(uint4*)&As[0][r][cc] = *(const uint4*)&A[(size_t)(rB + r) * K + k0 + cc];
      *(uint4*)&Bs[0][r][cc] = *(const uint4*)&BT[(size_t)(cB + r) * K + k0 + cc];
    }
    __syncthreads();
    Frag a[4], b[2];
#pragma unroll
    for (int i = 0; i < 4; ++i) {
      const ushort_t* p = &As[0][wr * 64 + i * 16 + mlane][kb8];
      a[i].u[0] = *(const uint4*)p;
      a[i].u[1] = *(const uint4*)(p + 16);
    }
#pragma unroll
    for (int j = 0; j < 2; ++j) {
      const ushort_t* p = &Bs[0][wc * 32 + j * 16 + mlane][kb8];
      b[j].u[0] = *(const uint4*)p;
      b[j].u[1] = *(const uint4*)(p + 16);
    }
#pragma unroll
    for (int i = 0; i < 4; ++i)
#pragma unroll
      for (int j = 0; j < 2; ++j)
        acc[i][j] = wmma_bf16(a[i].v, b[j].v, acc[i][j]);
    __syncthreads();
  }
#endif

  // Epilogue. C layout: col = lane&15, row = (lane>=16)*8 + vgpr_idx.
#pragma unroll
  for (int i = 0; i < 4; ++i) {
#pragma unroll
    for (int j = 0; j < 2; ++j) {
      int col = cB + wc * 32 + j * 16 + mlane;
      float bv = BIAS ? bias[col] : 0.f;
#pragma unroll
      for (int jj = 0; jj < 8; ++jj) {
        int row = rB + wr * 64 + i * 16 + kb8 + jj;
        float val = acc[i][j][jj] + bv;
        if (GELU) {
          float x3 = val * val * val;
          val = 0.5f * val *
                (1.f + tanhf(0.7978845608028654f * (val + 0.044715f * x3)));
        }
        size_t o = (size_t)row * Nc + col;
        if (OUTBF)      outB[o] = f2bf(val);
        else if (ACC)   outF[o] += val;
        else            outF[o] = val;
      }
    }
  }
  (void)M;
}

// ---------------------------------------------------------------------------
// Rotary + head reorder. qkv[N][6144] fp32 (MP=4 interleaved q|v|k blocks of
// 512) -> qh[h][n][d], kh[h][n][d] (rotary on d<64), vt[h][d][n], all bf16.
// ---------------------------------------------------------------------------
__global__ __launch_bounds__(256) void rot_kernel(const float* __restrict__ qkv,
                                                  ushort_t* __restrict__ qh,
                                                  ushort_t* __restrict__ kh,
                                                  ushort_t* __restrict__ vt) {
  int n = blockIdx.x;
  const float* row = qkv + (size_t)n * QKVC;
  for (int idx = threadIdx.x; idx < NH * DH; idx += 256) {
    int hh = idx >> 7, d = idx & 127;
    int base = (hh >> 2) * 1536 + (hh & 3) * 128;  // partition-major layout
    float qv = row[base + d];
    float vv = row[base + 512 + d];
    float kv = row[base + 1024 + d];
    if (d < DR) {
      int i = d >> 1;
      float invf = __powf(10000.f, -(float)(2 * i) * (1.f / 64.f));
      float ang = (float)n * invf;
      float sn = __sinf(ang), cs = __cosf(ang);
      float qp = row[base + (d ^ 1)];
      float kp = row[base + 1024 + (d ^ 1)];
      if ((d & 1) == 0) { qv = qv * cs - qp * sn; kv = kv * cs - kp * sn; }
      else              { qv = qv * cs + qp * sn; kv = kv * cs + kp * sn; }
    }
    qh[((size_t)hh * NTOK + n) * DH + d] = f2bf(qv);
    kh[((size_t)hh * NTOK + n) * DH + d] = f2bf(kv);
    vt[(size_t)hh * DH * NTOK + (size_t)d * NTOK + n] = f2bf(vv);
  }
}

// ---------------------------------------------------------------------------
// Flash attention (causal, online softmax). Grid: (N/128, H), 256 threads.
// Q cached in registers (4 A-fragments), K / V^T fragments straight from
// global (bf16, 16B aligned), P converted C-layout -> A-layout via LDS.
// ---------------------------------------------------------------------------
__global__ __launch_bounds__(256) void attn_kernel(
    const ushort_t* __restrict__ qh, const ushort_t* __restrict__ kh,
    const ushort_t* __restrict__ vt, ushort_t* __restrict__ av) {
  __shared__ alignas(16) ushort_t Pl[8][16][136];
  int h = blockIdx.y;
  int qb = blockIdx.x * 128;
  int tid = threadIdx.x, wave = tid >> 5, lane = tid & 31;
  int mlane = lane & 15, kb8 = (lane >> 4) * 8;
  int qrow0 = qb + wave * 16;
  const ushort_t* Qh = qh + (size_t)h * NTOK * DH;
  const ushort_t* Kh = kh + (size_t)h * NTOK * DH;
  const ushort_t* Vt = vt + (size_t)h * DH * NTOK;

  Frag qf[4];
#pragma unroll
  for (int kk = 0; kk < 4; ++kk) {
    const ushort_t* p = Qh + (size_t)(qrow0 + mlane) * DH + kk * 32 + kb8;
    qf[kk].u[0] = *(const uint4*)p;
    qf[kk].u[1] = *(const uint4*)(p + 16);
  }

  v8f o[8] = {};
  float mrun[8], lrun[8];
#pragma unroll
  for (int j = 0; j < 8; ++j) { mrun[j] = -3.0e38f; lrun[j] = 0.f; }

  int nchunks = blockIdx.x + 1;
  for (int ch = 0; ch < nchunks; ++ch) {
    int kc = ch * 128;
    v8f s[8] = {};
#pragma unroll
    for (int t = 0; t < 8; ++t) {
#pragma unroll
      for (int kk = 0; kk < 4; ++kk) {
        const ushort_t* p =
            Kh + (size_t)(kc + t * 16 + mlane) * DH + kk * 32 + kb8;
        Frag bf;
        bf.u[0] = *(const uint4*)p;
        bf.u[1] = *(const uint4*)(p + 16);
        s[t] = wmma_bf16(qf[kk].v, bf.v, s[t]);
      }
    }
    const float scl = 0.08838834764831845f;  // 1/sqrt(128)
    bool diag = (kc == qb);
    float mt[8];
#pragma unroll
    for (int j = 0; j < 8; ++j) mt[j] = -3.0e38f;
#pragma unroll
    for (int t = 0; t < 8; ++t) {
#pragma unroll
      for (int j = 0; j < 8; ++j) {
        float val = s[t][j] * scl;
        if (diag) {
          int r = qrow0 + kb8 + j;
          int c = kc + t * 16 + mlane;
          if (c > r) val = -1.0e10f;
        }
        s[t][j] = val;
        mt[j] = fmaxf(mt[j], val);
      }
    }
#pragma unroll
    for (int j = 0; j < 8; ++j) {
#pragma unroll
      for (int m = 1; m < 16; m <<= 1)
        mt[j] = fmaxf(mt[j], __shfl_xor(mt[j], m, 32));
    }
    float corr[8], rs[8];
#pragma unroll
    for (int j = 0; j < 8; ++j) {
      float mn = fmaxf(mrun[j], mt[j]);
      corr[j] = __expf(mrun[j] - mn);
      mrun[j] = mn;
      rs[j] = 0.f;
    }
#pragma unroll
    for (int t = 0; t < 8; ++t) {
#pragma unroll
      for (int j = 0; j < 8; ++j) {
        float pv = __expf(s[t][j] - mrun[j]);
        rs[j] += pv;
        Pl[wave][kb8 + j][t * 16 + mlane] = f2bf(pv);
      }
    }
#pragma unroll
    for (int j = 0; j < 8; ++j) {
#pragma unroll
      for (int m = 1; m < 16; m <<= 1) rs[j] += __shfl_xor(rs[j], m, 32);
      lrun[j] = lrun[j] * corr[j] + rs[j];
    }
#pragma unroll
    for (int t = 0; t < 8; ++t)
#pragma unroll
      for (int j = 0; j < 8; ++j) o[t][j] *= corr[j];
    __syncthreads();
#pragma unroll
    for (int kk = 0; kk < 4; ++kk) {
      Frag pf;
      const ushort_t* pp = &Pl[wave][mlane][kk * 32 + kb8];
      pf.u[0] = *(const uint4*)pp;
      pf.u[1] = *(const uint4*)(pp + 16);
#pragma unroll
      for (int t = 0; t < 8; ++t) {
        const ushort_t* vp =
            Vt + (size_t)(t * 16 + mlane) * NTOK + kc + kk * 32 + kb8;
        Frag bf;
        bf.u[0] = *(const uint4*)vp;
        bf.u[1] = *(const uint4*)(vp + 16);
        o[t] = wmma_bf16(pf.v, bf.v, o[t]);
      }
    }
    __syncthreads();
  }
#pragma unroll
  for (int j = 0; j < 8; ++j) lrun[j] = 1.f / lrun[j];
#pragma unroll
  for (int t = 0; t < 8; ++t) {
#pragma unroll
    for (int j = 0; j < 8; ++j) {
      int r = qrow0 + kb8 + j;
      int c = h * DH + t * 16 + mlane;
      av[(size_t)r * DMODEL + c] = f2bf(o[t][j] * lrun[j]);
    }
  }
}

// ---------------------------------------------------------------------------
extern "C" void kernel_launch(void* const* d_in, const int* in_sizes, int n_in,
                              void* d_out, int out_size, void* d_ws,
                              size_t ws_size, hipStream_t stream) {
  (void)in_sizes; (void)n_in; (void)out_size; (void)ws_size;
  const float* x     = (const float*)d_in[0];
  const float* ln_sc = (const float*)d_in[1];
  const float* ln_of = (const float*)d_in[2];
  const float* w_qkv = (const float*)d_in[3];
  const float* w_ao  = (const float*)d_in[4];
  const float* w_fi  = (const float*)d_in[5];
  const float* b_fi  = (const float*)d_in[6];
  const float* w_fo  = (const float*)d_in[7];
  const float* b_fo  = (const float*)d_in[8];
  float* out = (float*)d_out;

  char* ws = (char*)d_ws;
  size_t off = 0;
  auto take = [&](size_t bytes) -> char* {
    char* p = ws + off;
    off += (bytes + 255) & ~(size_t)255;
    return p;
  };
  ushort_t* hb    = (ushort_t*)take((size_t)NTOK * DMODEL * 2);
  ushort_t* wqkvT = (ushort_t*)take((size_t)QKVC * DMODEL * 2);
  ushort_t* waoT  = (ushort_t*)take((size_t)DMODEL * DMODEL * 2);
  ushort_t* wfiT  = (ushort_t*)take((size_t)FFD * DMODEL * 2);
  ushort_t* wfoT  = (ushort_t*)take((size_t)DMODEL * FFD * 2);
  float*    qkvf  = (float*)take((size_t)NTOK * QKVC * 4);
  ushort_t* qhB   = (ushort_t*)take((size_t)NH * NTOK * DH * 2);
  ushort_t* khB   = (ushort_t*)take((size_t)NH * NTOK * DH * 2);
  ushort_t* vtB   = (ushort_t*)take((size_t)NH * NTOK * DH * 2);
  ushort_t* avB   = (ushort_t*)take((size_t)NTOK * DMODEL * 2);
  // FF activation reuses the qkv fp32 buffer (dead after rot_kernel; 48MB>=32MB)
  ushort_t* act   = (ushort_t*)qkvf;

  ln_kernel<<<NTOK, 256, 0, stream>>>(x, ln_sc, ln_of, hb);
  wt_kernel<<<dim3(QKVC / 32, DMODEL / 32), 256, 0, stream>>>(w_qkv, wqkvT, DMODEL, QKVC);
  wt_kernel<<<dim3(DMODEL / 32, DMODEL / 32), 256, 0, stream>>>(w_ao, waoT, DMODEL, DMODEL);
  wt_kernel<<<dim3(FFD / 32, DMODEL / 32), 256, 0, stream>>>(w_fi, wfiT, DMODEL, FFD);
  wt_kernel<<<dim3(DMODEL / 32, FFD / 32), 256, 0, stream>>>(w_fo, wfoT, FFD, DMODEL);

  // QKV projection: [2048x2048] x [2048x6144] -> fp32
  gemm_kernel<0, 0, 0, 0><<<dim3(QKVC / 128, NTOK / 128), 256, 0, stream>>>(
      hb, wqkvT, nullptr, qkvf, nullptr, NTOK, QKVC, DMODEL);
  rot_kernel<<<NTOK, 256, 0, stream>>>(qkvf, qhB, khB, vtB);
  attn_kernel<<<dim3(NTOK / 128, NH), 256, 0, stream>>>(qhB, khB, vtB, avB);
  // attn output projection -> d_out (=)
  gemm_kernel<0, 0, 0, 0><<<dim3(DMODEL / 128, NTOK / 128), 256, 0, stream>>>(
      avB, waoT, nullptr, out, nullptr, NTOK, DMODEL, DMODEL);
  // FF in: bias + GELU -> bf16 activations
  gemm_kernel<1, 1, 1, 0><<<dim3(FFD / 128, NTOK / 128), 256, 0, stream>>>(
      hb, wfiT, b_fi, nullptr, act, NTOK, FFD, DMODEL);
  // FF out: bias, accumulate into d_out (+=)
  gemm_kernel<0, 1, 0, 1><<<dim3(DMODEL / 128, NTOK / 128), 256, 0, stream>>>(
      act, wfoT, b_fo, out, nullptr, NTOK, DMODEL, FFD);
}
